// VectorQuantizer_34608846471814
// MI455X (gfx1250) — matrix operand order repaired and checked by reading, not compiled, not run on hardware
//
#include <hip/hip_runtime.h>
#include <hip/hip_bf16.h>

typedef __attribute__((ext_vector_type(16))) _Float16 v16h;
typedef __attribute__((ext_vector_type(8)))  float    v8f;

static constexpr int NV  = 16384;  // number of input vectors (64*256)
static constexpr int DIM = 512;    // embedding dim
static constexpr int KCB = 8192;   // codebook entries

// ---------------- init ws accumulators ----------------
__global__ void vq_init(unsigned int* __restrict__ counts, float* __restrict__ loss_accum) {
    int i = blockIdx.x * blockDim.x + threadIdx.x;
    if (i < KCB) counts[i] = 0u;
    if (i == KCB) *loss_accum = 0.0f;
}

// ---------------- f32 -> f16 convert of inputs ----------------
__global__ void vq_cvt_inputs(const float* __restrict__ x, _Float16* __restrict__ a16) {
    size_t i = ((size_t)blockIdx.x * blockDim.x + threadIdx.x) * 4;
    float4 v = *(const float4*)(x + i);
    union { _Float16 h[4]; uint2 u; } r;
    r.h[0] = (_Float16)v.x; r.h[1] = (_Float16)v.y;
    r.h[2] = (_Float16)v.z; r.h[3] = (_Float16)v.w;
    *(uint2*)(a16 + i) = r.u;
}

// ---------------- f32 -> f16 codebook + row squared norms ----------------
__global__ void vq_cvt_codebook(const float* __restrict__ cb, _Float16* __restrict__ b16,
                                float* __restrict__ enorm) {
    __shared__ float sred[128];
    int row = blockIdx.x;
    int t = threadIdx.x;
    size_t base = (size_t)row * DIM + t * 4;
    float4 v = *(const float4*)(cb + base);
    union { _Float16 h[4]; uint2 u; } r;
    r.h[0] = (_Float16)v.x; r.h[1] = (_Float16)v.y;
    r.h[2] = (_Float16)v.z; r.h[3] = (_Float16)v.w;
    *(uint2*)(b16 + base) = r.u;
    sred[t] = v.x*v.x + v.y*v.y + v.z*v.z + v.w*v.w;
    __syncthreads();
    for (int off = 64; off > 0; off >>= 1) {
        if (t < off) sred[t] += sred[t + off];
        __syncthreads();
    }
    if (t == 0) enorm[row] = sred[0];
}

// ---------------- WMMA distance + argmin (async double-buffered LDS) ----------------
// Block: 256 threads = 8 waves; block owns 128 rows, wave owns a 16-row tile.
// A (16x32 f16 per chunk) preloaded into VGPRs; codebook panels (16 cols x 512 f16 = 16 KB)
// double-buffered in LDS and filled via CDNA5 async global->LDS DMA (ASYNCcnt-tracked),
// so the L2 fetch of panel i+1 overlaps the 16 WMMAs of panel i.
__global__ void __launch_bounds__(256) vq_argmin(const _Float16* __restrict__ a16,
                                                 const _Float16* __restrict__ b16,
                                                 const float* __restrict__ enorm,
                                                 int* __restrict__ outIdx) {
    __shared__ __align__(16) _Float16 bsh[2][16 * DIM];  // 2 x 16 KB panels
    const int t    = threadIdx.x;
    const int wave = t >> 5;
    const int lane = t & 31;
    const int half = lane >> 4;   // 0: lanes 0-15, 1: lanes 16-31
    const int lcol = lane & 15;
    const int rowBase = blockIdx.x * 128 + wave * 16;

    // --- preload A: 16 K-chunks of the 16x32 f16 tile (ISA A-layout, interleaved) ---
    // lane half h holds K = [8h..8h+7] in halves 0..7 and K = [16+8h..16+8h+7] in halves 8..15
    v16h a[16];
    {
        const _Float16* ap = a16 + (size_t)(rowBase + lcol) * DIM + half * 8;
        #pragma unroll
        for (int kc = 0; kc < 16; ++kc) {
            union { v16h v; uint4 q[2]; } u;
            u.q[0] = *(const uint4*)(ap + kc * 32);
            u.q[1] = *(const uint4*)(ap + kc * 32 + 16);
            a[kc] = u.v;
        }
    }

    // per-thread async copy slots: 4 x 16B, stride 4 KB (256 threads cover 16 KB)
    const unsigned lds0 = (unsigned)(size_t)(&bsh[0][0]) + (unsigned)t * 16u;
    const unsigned lds1 = (unsigned)(size_t)(&bsh[1][0]) + (unsigned)t * 16u;
    const unsigned long long gbase = (unsigned long long)(size_t)b16 + (unsigned long long)t * 16u;

    // prefetch panel 0 into buffer 0 (async, no wait yet)
    #pragma unroll
    for (int i = 0; i < 4; ++i) {
        asm volatile("global_load_async_to_lds_b128 %0, %1, off"
                     :: "v"(lds0 + (unsigned)i * 4096u),
                        "v"(gbase + (unsigned long long)i * 4096ull)
                     : "memory");
    }

    float bestv[8];
    int   besti[8];
    #pragma unroll
    for (int j = 0; j < 8; ++j) { bestv[j] = 3.4e38f; besti[j] = 0; }

    for (int cbk = 0; cbk < KCB / 16; ++cbk) {
        const int colBase = cbk * 16;
        const _Float16* bufCur = bsh[cbk & 1];

        if (cbk + 1 < KCB / 16) {
            // issue async copies for the NEXT panel into the other buffer
            const unsigned ldsNxt = (cbk & 1) ? lds0 : lds1;
            const unsigned long long gnxt =
                gbase + (unsigned long long)(colBase + 16) * DIM * 2ull;
            #pragma unroll
            for (int i = 0; i < 4; ++i) {
                asm volatile("global_load_async_to_lds_b128 %0, %1, off"
                             :: "v"(ldsNxt + (unsigned)i * 4096u),
                                "v"(gnxt + (unsigned long long)i * 4096ull)
                             : "memory");
            }
            // current panel's 4 asyncs (issued last iteration) are done when cnt <= 4
            asm volatile("s_wait_asynccnt 0x4" ::: "memory");
        } else {
            asm volatile("s_wait_asynccnt 0x0" ::: "memory");
        }
        __syncthreads();  // current panel visible to all waves

        v8f c = {0.f, 0.f, 0.f, 0.f, 0.f, 0.f, 0.f, 0.f};
        // B layout (matches A): lane half h holds K = [8h..8h+7] in halves 0..7 and K = [16+8h..16+8h+7] in halves 8..15
        const _Float16* bp = bufCur + lcol * DIM + half * 8;
        #pragma unroll
        for (int kc = 0; kc < 16; ++kc) {
            union { v16h v; uint4 q[2]; } u;
            u.q[0] = *(const uint4*)(bp + kc * 32);
            u.q[1] = *(const uint4*)(bp + kc * 32 + 16);
            c = __builtin_amdgcn_wmma_f32_16x16x32_f16(false, a[kc], false, u.v,
                                                       (short)0, c, false, false);
        }

        // score = ||e||^2 - 2 * dot ; C layout: elem j, lane L -> (M=j+8*half, N=lcol)
        const int   n  = colBase + lcol;
        const float en = enorm[n];
        #pragma unroll
        for (int j = 0; j < 8; ++j) {
            float s = en - 2.0f * c[j];
            if (s < bestv[j]) { bestv[j] = s; besti[j] = n; }
        }
        __syncthreads();  // all waves done reading this panel before it is overwritten
    }

    // reduce over the 16 lanes of each half (same row, different columns)
    #pragma unroll
    for (int j = 0; j < 8; ++j) {
        float v = bestv[j];
        int   bi = besti[j];
        #pragma unroll
        for (int off = 1; off < 16; off <<= 1) {
            float ov = __shfl_xor(v, off, 32);
            int   oi = __shfl_xor(bi, off, 32);
            if (ov < v || (ov == v && oi < bi)) { v = ov; bi = oi; }
        }
        besti[j] = bi;
    }
    if (lcol == 0) {
        #pragma unroll
        for (int j = 0; j < 8; ++j)
            outIdx[rowBase + half * 8 + j] = besti[j];
    }
}

// ---------------- gather quantized, loss partials, histogram ----------------
__global__ void vq_quant(const float* __restrict__ x, const float* __restrict__ cbk,
                         const int* __restrict__ idx, float* __restrict__ qout,
                         float* __restrict__ loss_accum, unsigned int* __restrict__ counts) {
    __shared__ float sred[256];
    int row = blockIdx.x;
    int t = threadIdx.x;
    int k = idx[row];
    float s = 0.f;
    #pragma unroll
    for (int i = 0; i < 2; ++i) {
        int c = t + i * 256;
        float e  = cbk[(size_t)k * DIM + c];
        float xv = x[(size_t)row * DIM + c];
        qout[(size_t)row * DIM + c] = e;   // quantized_st == quantized numerically
        float d = e - xv;
        s += d * d;
    }
    sred[t] = s;
    __syncthreads();
    for (int off = 128; off > 0; off >>= 1) {
        if (t < off) sred[t] += sred[t + off];
        __syncthreads();
    }
    if (t == 0) {
        atomicAdd(loss_accum, sred[0]);
        atomicAdd(&counts[k], 1u);
    }
}

// ---------------- one-hot encodings (zero + one fused), float4 stores ----------------
__global__ void vq_encodings(const int* __restrict__ idx, float* __restrict__ enc) {
    size_t i = (size_t)blockIdx.x * blockDim.x + threadIdx.x;  // float4 index
    int row  = (int)(i >> 11);               // (i*4) / 8192
    int col0 = (int)((i & 2047) << 2);
    int k = idx[row];
    float4 v;
    v.x = (col0     == k) ? 1.f : 0.f;
    v.y = (col0 + 1 == k) ? 1.f : 0.f;
    v.z = (col0 + 2 == k) ? 1.f : 0.f;
    v.w = (col0 + 3 == k) ? 1.f : 0.f;
    ((float4*)enc)[i] = v;
}

// ---------------- loss + perplexity ----------------
__global__ void vq_finalize(const float* __restrict__ loss_accum,
                            const unsigned int* __restrict__ counts,
                            float* __restrict__ out_loss, float* __restrict__ out_ppl) {
    __shared__ float sred[256];
    int t = threadIdx.x;
    float h = 0.f;
    for (int k = t; k < KCB; k += 256) {
        float p = (float)counts[k] * (1.0f / (float)NV);
        h += p * logf(p + 1e-10f);
    }
    sred[t] = h;
    __syncthreads();
    for (int off = 128; off > 0; off >>= 1) {
        if (t < off) sred[t] += sred[t + off];
        __syncthreads();
    }
    if (t == 0) {
        *out_ppl  = expf(-sred[0]);
        *out_loss = (*loss_accum) * (1.25f / (float)((size_t)NV * DIM));
    }
}

extern "C" void kernel_launch(void* const* d_in, const int* in_sizes, int n_in,
                              void* d_out, int out_size, void* d_ws, size_t ws_size,
                              hipStream_t stream) {
    const float* x   = (const float*)d_in[0];  // [64,256,512] f32
    const float* cbk = (const float*)d_in[1];  // [8192,512]   f32
    float* out = (float*)d_out;

    // workspace layout (~25.3 MB)
    char* ws = (char*)d_ws;
    _Float16* A16 = (_Float16*)ws;                                   // 16 MB
    _Float16* B16 = (_Float16*)(ws + (size_t)NV * DIM * 2);          //  8 MB
    float*    enorm = (float*)(ws + (size_t)NV * DIM * 2 + (size_t)KCB * DIM * 2);
    int*      idx   = (int*)((char*)enorm + (size_t)KCB * 4);
    unsigned* counts = (unsigned*)((char*)idx + (size_t)NV * 4);
    float*    loss_accum = (float*)((char*)counts + (size_t)KCB * 4);

    // output layout: loss(1) | quantized_st(NV*DIM) | perplexity(1) | encodings(NV*KCB)
    float* out_loss = out;
    float* out_q    = out + 1;
    float* out_ppl  = out + 1 + (size_t)NV * DIM;
    float* out_enc  = out + 2 + (size_t)NV * DIM;

    vq_init<<<(KCB + 256) / 256, 256, 0, stream>>>(counts, loss_accum);
    vq_cvt_inputs<<<(NV * DIM / 4) / 256, 256, 0, stream>>>(x, A16);
    vq_cvt_codebook<<<KCB, 128, 0, stream>>>(cbk, B16, enorm);
    vq_argmin<<<NV / 128, 256, 0, stream>>>(A16, B16, enorm, idx);
    vq_quant<<<NV, 256, 0, stream>>>(x, cbk, idx, out_q, loss_accum, counts);
    vq_encodings<<<(unsigned)(((size_t)NV * KCB / 4) / 256), 256, 0, stream>>>(idx, out_enc);
    vq_finalize<<<1, 256, 0, stream>>>(loss_accum, counts, out_loss, out_ppl);
}
